// NeuralBondOrder_47425028883062
// MI455X (gfx1250) — compile-verified, matrix operand order checked by reading, not run on hardware
//
#include <hip/hip_runtime.h>
#include <hip/hip_bf16.h>

// ---------------------------------------------------------------------------
// ALIGNN-style GNN forward + pair-potential energy/forces for MI455X (gfx1250)
// [*,64]x[64,64] GEMMs on v_wmma_f32_16x16x32_bf16 (wave32 WMMA).
// Weights fetched once per block via the Tensor Data Mover (tensor_load_to_lds,
// TENSORcnt), swizzled in LDS into B-fragment layout, then held in VGPRs while
// rows stream through the matrix pipe.
// ---------------------------------------------------------------------------

#define BN_SCALE 0.9999950000374997f   // 1/sqrt(1+1e-5)
#define PI_F 3.14159265358979f

typedef __attribute__((ext_vector_type(16))) __bf16 v16bf;
typedef __attribute__((ext_vector_type(8)))  float  v8f;
typedef __attribute__((ext_vector_type(4)))  unsigned su4;
typedef __attribute__((ext_vector_type(8)))  unsigned su8;

union Frag16 { unsigned short u[16]; v16bf v; };

__device__ __forceinline__ unsigned short f2bf(float f) {
  // round-to-nearest-even f32 -> bf16
  unsigned u = __float_as_uint(f);
  u += 0x7fffu + ((u >> 16) & 1u);
  return (unsigned short)(u >> 16);
}

// K-index inside a 16x32 (or 32x16) bf16 WMMA fragment for a given lane/slot.
// Per CDNA5 ISA 7.12.2: lanes 0-15 hold K 0..7 (vgpr0-3) and 16..23 (vgpr4-7);
// lanes 16-31 hold K 8..15 and 24..31.
__device__ __forceinline__ int wmma_kmap(int lane, int i) {
  int vg = i >> 1, h = i & 1;
  int base = (vg < 4) ? (2 * vg) : (16 + 2 * (vg - 4));
  return base + h + ((lane >= 16) ? 8 : 0);
}

__device__ __forceinline__ float fast_sigmoid(float v) {
  return __builtin_amdgcn_rcpf(1.f + __expf(-v));   // v_rcp_f32, no IEEE div
}
__device__ __forceinline__ float fast_silu(float v) {
  return v * fast_sigmoid(v);
}

// ---------------------------------------------------------------------------
// Fused row-streaming GEMM: out[M,64] = act( A[M,Kp] @ W[K,64] + b )
// NKT = Kp/32 (compile-time). ACT: BN scale + SiLU epilogue (compile-time).
// W is DMA'd into LDS by the Tensor Data Mover (one wave), swizzled into WMMA
// B-fragment layout; the compiler keeps the fragments in VGPRs across rows.
// ---------------------------------------------------------------------------
template <int NKT, int ACT>
__global__ void __launch_bounds__(256)
gemm64_wmma(const float* __restrict__ A, int M, int K,
            const float* __restrict__ W, const float* __restrict__ bias,
            float* __restrict__ out)
{
  constexpr int Kp = NKT * 32;
  __shared__ float wraw[96 * 64];                   // raw W tile (TDM target), 24KB
  __shared__ unsigned short wfrag[3 * 4 * 32 * 16]; // B-fragment layout, 12KB
  const int tid  = threadIdx.x;
  const int wave = tid >> 5;
  const int lane = tid & 31;

  // ---- TDM bulk copy: W[K,64] f32 -> LDS (wave 0 issues, EXEC-independent) ----
  if (wave == 0) {
    unsigned long long wp = (unsigned long long)W;
    unsigned ldsoff = (unsigned)(size_t)(void*)wraw; // low 32 bits = LDS offset
    su4 g0;
    g0[0] = 1u;                 // count=1 valid descriptor
    g0[1] = ldsoff;             // lds_addr
    g0[2] = (unsigned)wp;       // global_addr[31:0]
    g0[3] = ((unsigned)(wp >> 32) & 0x01FFFFFFu) | (2u << 30); // addr[56:32] | type=2
    su8 g1;
    g1[0] = 0x20000u;           // data_size=4B (bits 17:16 = 2), no multicast
    g1[1] = (64u << 16);        // tensor_dim0 = 64 (bits 79:48, low part)
    g1[2] = ((unsigned)K << 16);// tensor_dim1 = K rows (bits 111:80, low part)
    g1[3] = (64u << 16);        // tile_dim0 = 64 (bits 127:112)
    g1[4] = (unsigned)K;        // tile_dim1 = K (bits 143:128)
    g1[5] = 64u;                // tensor_dim0_stride = 64 (bits 207:160, low)
    g1[6] = 0u;
    g1[7] = 0u;
    asm volatile("tensor_load_to_lds %0, %1" :: "s"(g0), "s"(g1) : "memory");
    __builtin_amdgcn_s_wait_tensorcnt(0);
  }
  __syncthreads();

  // ---- swizzle LDS raw W into B-fragment layout (bf16) ----
  for (int idx = tid; idx < NKT * 4 * 32 * 16; idx += 256) {
    int i  = idx & 15;
    int ln = (idx >> 4) & 31;
    int nt = (idx >> 9) & 3;
    int kt = idx >> 11;
    int k = kt * 32 + wmma_kmap(ln, i);
    int n = nt * 16 + (ln & 15);
    wfrag[idx] = (k < K) ? f2bf(wraw[k * 64 + n]) : (unsigned short)0;
  }
  __syncthreads();

  const int hi8   = (lane >= 16) ? 8 : 0;
  const int mstep = gridDim.x * 8 * 16;

  for (int r0 = (blockIdx.x * 8 + wave) * 16; r0 < M; r0 += mstep) {
    const int row = r0 + (lane & 15);
    const bool rv = (row < M);

    // Load + convert A fragments (two contiguous f32 per slot pair).
    Frag16 afr[NKT];
    #pragma unroll
    for (int kt = 0; kt < NKT; ++kt) {
      const float* ap = A + (size_t)row * Kp + kt * 32 + hi8;
      #pragma unroll
      for (int vg = 0; vg < 8; ++vg) {
        int base = (vg < 4) ? (2 * vg) : (16 + 2 * (vg - 4));
        float f0 = rv ? ap[base]     : 0.f;
        float f1 = rv ? ap[base + 1] : 0.f;
        afr[kt].u[2 * vg]     = f2bf(f0);
        afr[kt].u[2 * vg + 1] = f2bf(f1);
      }
    }

    #pragma unroll
    for (int nt = 0; nt < 4; ++nt) {
      v8f acc = {0.f, 0.f, 0.f, 0.f, 0.f, 0.f, 0.f, 0.f};
      #pragma unroll
      for (int kt = 0; kt < NKT; ++kt) {
        Frag16 bfr;
        const unsigned short* wp2 = &wfrag[((((kt << 2) | nt) * 32) + lane) * 16];
        #pragma unroll
        for (int i = 0; i < 16; ++i) bfr.u[i] = wp2[i];
        acc = __builtin_amdgcn_wmma_f32_16x16x32_bf16(
                  false, afr[kt].v, false, bfr.v, (short)0, acc, false, false);
      }
      const int n = (nt << 4) | (lane & 15);
      const float bv = bias[n];
      #pragma unroll
      for (int vg = 0; vg < 8; ++vg) {
        int orow = r0 + hi8 + vg;          // C layout: M = vgpr + 8*(lane>=16)
        if (orow < M) {
          float val = acc[vg] + bv;
          if (ACT) { val *= BN_SCALE; val = fast_silu(val); }
          out[(size_t)orow * 64 + n] = val;
        }
      }
    }
  }
}

// ---------------------------------------------------------------------------
// Elementwise / gather-scatter kernels
// ---------------------------------------------------------------------------
__global__ void zero_kernel(float* __restrict__ p, size_t n) {
  size_t i = (size_t)blockIdx.x * blockDim.x + threadIdx.x;
  for (; i < n; i += (size_t)gridDim.x * blockDim.x) p[i] = 0.f;
}

__global__ void embed_kernel(const float* __restrict__ emb, const int* __restrict__ z,
                             float* __restrict__ x, float* __restrict__ x0, int Nn) {
  int i = blockIdx.x * blockDim.x + threadIdx.x;
  if (i >= Nn * 64) return;
  int n = i >> 6, c = i & 63;
  float v = emb[(size_t)z[n] * 64 + c];
  x[i] = v; x0[i] = v;
}

__global__ void bond_kernel(const float* __restrict__ r, float* __restrict__ bl,
                            float* __restrict__ ibl, int E) {
  int e = blockIdx.x * blockDim.x + threadIdx.x;
  if (e >= E) return;
  float x = r[3 * e], y = r[3 * e + 1], z = r[3 * e + 2];
  float b = sqrtf(x * x + y * y + z * z);
  bl[e] = b; ibl[e] = __builtin_amdgcn_rcpf(b);
}

__global__ void rbf_edge_kernel(const float* __restrict__ bl, float* __restrict__ out, int E) {
  size_t i = (size_t)blockIdx.x * blockDim.x + threadIdx.x;
  if (i >= (size_t)E * 96) return;
  int e = (int)(i / 96), k = (int)(i % 96);
  float v = 0.f;
  if (k < 80) {
    float c = 8.f * (float)k / 79.f;
    float d = bl[e] - c;
    v = __expf(-(79.f / 8.f) * d * d);
  }
  out[i] = v;
}

__global__ void cos_kernel(const float* __restrict__ r, const float* __restrict__ ibl,
                           const int* __restrict__ ls, const int* __restrict__ ld,
                           float* __restrict__ cosv, int L) {
  int l = blockIdx.x * blockDim.x + threadIdx.x;
  if (l >= L) return;
  int a = ls[l], b = ld[l];
  // r1 = -r[a], r2 = r[b]
  float dot = -(r[3 * a] * r[3 * b] + r[3 * a + 1] * r[3 * b + 1] + r[3 * a + 2] * r[3 * b + 2]);
  float c = dot * ibl[a] * ibl[b];
  cosv[l] = fminf(1.f, fmaxf(-1.f, c));
}

__global__ void rbf_ang_kernel(const float* __restrict__ cosv, float* __restrict__ out, int L) {
  size_t i = (size_t)blockIdx.x * blockDim.x + threadIdx.x;
  if (i >= (size_t)L * 64) return;
  int l = (int)(i >> 6), k = (int)(i & 63);
  float v = 0.f;
  if (k < 40) {
    float c = -1.f + 2.f * (float)k / 39.f;
    float d = cosv[l] - c;
    v = __expf(-(39.f / 2.f) * d * d);
  }
  out[i] = v;
}

// EGGC edge stage: m = u0[src]+u1[dst]+v2[e]; sigma=sigmoid(m);
// scatter sigma*Bh and sigma to dst segments; y <- y + silu(m*BN).
__global__ void eggc_edge_kernel(const int* __restrict__ es, const int* __restrict__ ed, int E,
                                 const float* __restrict__ u0, const float* __restrict__ u1,
                                 const float* __restrict__ v2, const float* __restrict__ u4,
                                 const float* __restrict__ yin, float* __restrict__ yout,
                                 float* __restrict__ ssh, float* __restrict__ ss) {
  size_t i = (size_t)blockIdx.x * blockDim.x + threadIdx.x;
  if (i >= (size_t)E * 64) return;
  int e = (int)(i >> 6), c = (int)(i & 63);
  int s = es[e], d = ed[e];
  __builtin_prefetch(&u0[(size_t)s * 64 + c], 0, 1);   // global_prefetch_b8
  float m = u0[(size_t)s * 64 + c] + u1[(size_t)d * 64 + c] + v2[i];
  float sg = fast_sigmoid(m);
  float bh = u4[(size_t)s * 64 + c];
  atomicAdd(&ssh[(size_t)d * 64 + c], sg * bh);
  atomicAdd(&ss[(size_t)d * 64 + c], sg);
  float yv = yin[i];
  yout[i] = yv + fast_silu(m * BN_SCALE);
}

// EGGC node stage: h = ssh/(ss+1e-6); x <- x + silu((u3+h)*BN).
__global__ void eggc_node_kernel(int Nn, const float* __restrict__ ssh,
                                 const float* __restrict__ ss, const float* __restrict__ u3,
                                 const float* __restrict__ xin, float* __restrict__ xout) {
  size_t i = (size_t)blockIdx.x * blockDim.x + threadIdx.x;
  if (i >= (size_t)Nn * 64) return;
  float h = ssh[i] * __builtin_amdgcn_rcpf(ss[i] + 1e-6f);
  xout[i] = xin[i] + fast_silu((u3[i] + h) * BN_SCALE);
}

__global__ void fc_kernel(const float* __restrict__ y, const float* __restrict__ fcW,
                          const float* __restrict__ fcb, float* __restrict__ bo, int E) {
  int e = blockIdx.x * blockDim.x + threadIdx.x;
  if (e >= E) return;
  const float* yr = y + (size_t)e * 64;
  __builtin_prefetch(yr, 0, 1);
  float acc = fcb[0];
  #pragma unroll 8
  for (int c = 0; c < 64; ++c) acc += yr[c] * fcW[c];
  bo[e] = fast_sigmoid(acc);
}

__global__ void pairfeat_kernel(const float* __restrict__ x0,
                                const float* __restrict__ Wsrc, const float* __restrict__ bsrc,
                                const float* __restrict__ Wdst,
                                float* __restrict__ ps, float* __restrict__ pd, int Nn) {
  int i = blockIdx.x * blockDim.x + threadIdx.x;
  if (i >= Nn * 4) return;
  int n = i >> 2, j = i & 3;
  const float* xr = x0 + (size_t)n * 64;
  float a = bsrc[j], b = 0.f;
  #pragma unroll 8
  for (int c = 0; c < 64; ++c) {
    float xv = xr[c];
    a += xv * Wsrc[c * 4 + j];
    b += xv * Wdst[c * 4 + j];
  }
  ps[i] = a; pd[i] = b;
}

// Pair potential: V_pair scatter, plus direct dV/d|r| force path scattered to dst.
__global__ void pair_kernel(const float* __restrict__ r, const float* __restrict__ bl,
                            const float* __restrict__ ibl,
                            const int* __restrict__ src, const int* __restrict__ dst,
                            const float* __restrict__ ps, const float* __restrict__ pd,
                            const float* __restrict__ bo,
                            float* __restrict__ V, float* __restrict__ outv, int E) {
  int e = blockIdx.x * blockDim.x + threadIdx.x;
  if (e >= E) return;
  int s = src[e], d = dst[e];
  float p0 = __expf(ps[4 * s + 0] + pd[4 * d + 0]);
  float p1 = __expf(ps[4 * s + 1] + pd[4 * d + 1]);
  float p2 = __expf(ps[4 * s + 2] + pd[4 * d + 2]);
  float p3 = __expf(ps[4 * s + 3] + pd[4 * d + 3]);
  float b = bl[e];
  float frep = p0 * __expf(-p1 * b);
  float fatt = p2 * __expf(-p3 * b);
  float boe = bo[e];
  const float D = 0.1f, R = 3.9f;
  float c, dc;
  if (b < R - D)      { c = 1.f; dc = 0.f; }
  else if (b > R + D) { c = 0.f; dc = 0.f; }
  else {
    float t = PI_F * (b - R) / (2.f * D);
    c  = 0.5f - 0.5f * __sinf(t);
    dc = -0.5f * __cosf(t) * PI_F / (2.f * D);
  }
  float g = frep - boe * fatt;
  atomicAdd(&V[d], c * g);
  // dV/dbl (direct pair path); forces = -dV/dbl * r_hat scattered to dst
  float dV = dc * g + c * (-p1 * frep + boe * p3 * fatt);
  float sc = -dV * ibl[e];
  atomicAdd(&outv[1 + 3 * d + 0], sc * r[3 * e + 0]);
  atomicAdd(&outv[1 + 3 * d + 1], sc * r[3 * e + 1]);
  atomicAdd(&outv[1 + 3 * d + 2], sc * r[3 * e + 2]);
}

__global__ void energy_kernel(const float* __restrict__ V, float* __restrict__ out, int Nn) {
  int n = blockIdx.x * blockDim.x + threadIdx.x;
  if (n >= Nn) return;
  atomicAdd(out, V[n] * (1.f / (float)Nn));
}

// ---------------------------------------------------------------------------
// Host orchestration
// ---------------------------------------------------------------------------
static inline int g1(size_t n) {
  size_t b = (n + 255) / 256;
  return (int)(b < 1 ? 1 : b);
}

static inline void launch_gemm(hipStream_t st, const float* A, int M, int K, int Kp,
                               const float* W, const float* bias, float* out, int act) {
  int rowBlocks = (M + 15) / 16;
  int blocks = (rowBlocks + 7) / 8;
  if (blocks > 2048) blocks = 2048;
  if (blocks < 1) blocks = 1;
  if (Kp == 96) {
    gemm64_wmma<3, 1><<<blocks, 256, 0, st>>>(A, M, K, W, bias, out); // only ACT users
  } else if (act) {
    gemm64_wmma<2, 1><<<blocks, 256, 0, st>>>(A, M, K, W, bias, out);
  } else {
    gemm64_wmma<2, 0><<<blocks, 256, 0, st>>>(A, M, K, W, bias, out);
  }
}

extern "C" void kernel_launch(void* const* d_in, const int* in_sizes, int n_in,
                              void* d_out, int out_size, void* d_ws, size_t ws_size,
                              hipStream_t stream) {
  const float* r        = (const float*)d_in[0];
  const float* atom_emb = (const float*)d_in[1];
  const float* eW1 = (const float*)d_in[2];  const float* eb1 = (const float*)d_in[3];
  const float* eW2 = (const float*)d_in[4];  const float* eb2 = (const float*)d_in[5];
  const float* aW1 = (const float*)d_in[6];  const float* ab1 = (const float*)d_in[7];
  const float* aW2 = (const float*)d_in[8];  const float* ab2 = (const float*)d_in[9];
  const float* gW  = (const float*)d_in[10]; const float* gB  = (const float*)d_in[11];
  const float* fcW = (const float*)d_in[12]; const float* fcb = (const float*)d_in[13];
  const float* pWs = (const float*)d_in[14]; const float* pbs = (const float*)d_in[15];
  const float* pWd = (const float*)d_in[16];
  const int* atom_z = (const int*)d_in[17];
  const int* src    = (const int*)d_in[18];
  const int* dst    = (const int*)d_in[19];
  const int* lg_src = (const int*)d_in[20];
  const int* lg_dst = (const int*)d_in[21];

  const int Nn = in_sizes[17];
  const int E  = in_sizes[18];
  const int L  = in_sizes[20];
  float* out = (float*)d_out;

  // ---- workspace carve ----
  float* p = (float*)d_ws;
  auto take = [&](size_t n) { float* q = p; p += n; return q; };
  float* bl   = take((size_t)E);
  float* ibl  = take((size_t)E);
  float* rbfe = take((size_t)E * 96);
  float* ytmp = take((size_t)E * 64);
  float* y    = take((size_t)E * 64);
  float* cosv = take((size_t)L);
  float* rbfa = take((size_t)L * 64);
  float* z    = take((size_t)L * 64);
  float* tlg  = take((size_t)L * 64);   // v2 buffer for line-graph convs, also ang-MLP temp
  float* x    = take((size_t)Nn * 64);
  float* x0   = take((size_t)Nn * 64);
  float* u0   = take((size_t)E * 64);
  float* u1   = take((size_t)E * 64);
  float* u3   = take((size_t)E * 64);
  float* u4   = take((size_t)E * 64);
  float* v2g  = take((size_t)E * 64);   // v2 buffer for atom-graph convs
  float* ssh  = take((size_t)E * 64);
  float* ss   = take((size_t)E * 64);
  float* bo   = take((size_t)E);
  float* ps   = take((size_t)Nn * 4);
  float* pd   = take((size_t)Nn * 4);
  float* V    = take((size_t)Nn);
  (void)ws_size; (void)n_in; (void)out_size;

  // ---- init outputs ----
  zero_kernel<<<g1((size_t)1 + 3 * Nn), 256, 0, stream>>>(out, (size_t)1 + 3 * Nn);

  // ---- embeddings & geometry ----
  embed_kernel<<<g1((size_t)Nn * 64), 256, 0, stream>>>(atom_emb, atom_z, x, x0, Nn);
  bond_kernel<<<g1((size_t)E), 256, 0, stream>>>(r, bl, ibl, E);

  // ---- edge feature MLP: y = mlp(mlp(rbf(bl,0,8,80))) ----
  rbf_edge_kernel<<<g1((size_t)E * 96), 256, 0, stream>>>(bl, rbfe, E);
  launch_gemm(stream, rbfe, E, 80, 96, eW1, eb1, ytmp, 1);
  launch_gemm(stream, ytmp, E, 64, 64, eW2, eb2, y, 1);

  // ---- angle feature MLP: z = mlp(mlp(rbf(cos,-1,1,40))) ----
  cos_kernel<<<g1((size_t)L), 256, 0, stream>>>(r, ibl, lg_src, lg_dst, cosv, L);
  rbf_ang_kernel<<<g1((size_t)L * 64), 256, 0, stream>>>(cosv, rbfa, L);
  launch_gemm(stream, rbfa, L, 40, 64, aW1, ab1, tlg, 1);
  launch_gemm(stream, tlg,  L, 64, 64, aW2, ab2, z, 1);

  // ---- EGGC layer driver ----
  auto run_eggc = [&](int layer, const int* es, const int* ed, int nNode, int nE,
                      float* xf, float* yf, float* v2buf) {
    const float* Wb = gW + (size_t)layer * 5 * 4096;
    const float* bb = gB + (size_t)layer * 5 * 64;
    launch_gemm(stream, xf, nNode, 64, 64, Wb + 0 * 4096, bb + 0 * 64, u0, 0);
    launch_gemm(stream, xf, nNode, 64, 64, Wb + 1 * 4096, bb + 1 * 64, u1, 0);
    launch_gemm(stream, xf, nNode, 64, 64, Wb + 3 * 4096, bb + 3 * 64, u3, 0);
    launch_gemm(stream, xf, nNode, 64, 64, Wb + 4 * 4096, bb + 4 * 64, u4, 0);
    launch_gemm(stream, yf, nE,    64, 64, Wb + 2 * 4096, bb + 2 * 64, v2buf, 0);
    zero_kernel<<<g1((size_t)nNode * 64), 256, 0, stream>>>(ssh, (size_t)nNode * 64);
    zero_kernel<<<g1((size_t)nNode * 64), 256, 0, stream>>>(ss, (size_t)nNode * 64);
    eggc_edge_kernel<<<g1((size_t)nE * 64), 256, 0, stream>>>(es, ed, nE, u0, u1, v2buf,
                                                              u4, yf, yf, ssh, ss);
    eggc_node_kernel<<<g1((size_t)nNode * 64), 256, 0, stream>>>(nNode, ssh, ss, u3, xf, xf);
  };

  // ALIGNN layers (atom-graph conv then line-graph conv), x2
  run_eggc(0, src,    dst,    Nn, E, x, y, v2g);
  run_eggc(1, lg_src, lg_dst, E,  L, y, z, tlg);
  run_eggc(2, src,    dst,    Nn, E, x, y, v2g);
  run_eggc(3, lg_src, lg_dst, E,  L, y, z, tlg);
  // GCN layers, x2
  run_eggc(4, src,    dst,    Nn, E, x, y, v2g);
  run_eggc(5, src,    dst,    Nn, E, x, y, v2g);

  // ---- bond order + pair potential + energy + forces ----
  fc_kernel<<<g1((size_t)E), 256, 0, stream>>>(y, fcW, fcb, bo, E);
  pairfeat_kernel<<<g1((size_t)Nn * 4), 256, 0, stream>>>(x0, pWs, pbs, pWd, ps, pd, Nn);
  zero_kernel<<<g1((size_t)Nn), 256, 0, stream>>>(V, (size_t)Nn);
  pair_kernel<<<g1((size_t)E), 256, 0, stream>>>(r, bl, ibl, src, dst, ps, pd, bo, V, out, E);
  energy_kernel<<<g1((size_t)Nn), 256, 0, stream>>>(V, out, Nn);
}
